// MultiQueryAttention_80968723464445
// MI455X (gfx1250) — matrix operand order repaired and checked by reading, not compiled, not run on hardware
//
#include <hip/hip_runtime.h>

// ---------------------------------------------------------------------------
// MultiQueryAttention (GQA) forward for gfx1250 (MI455X), bf16 WMMA pipeline.
//   B=2, S=2048, D_MODEL=2048, NUM_HEADS=32, NUM_KV_HEADS=8, HEAD_DIM=64
// Pipeline: [x@Wq, x@Wk, x@Wv](wmma) -> RoPE(valu) -> flash-attn(wmma) -> @Wo(wmma)
// Data movement: TDM tensor_load_to_lds / global_load_async_to_lds_b128 where
// the toolchain exposes them (guarded), else plain global->VGPR->LDS staging.
// ---------------------------------------------------------------------------

#define S_LEN 2048
#define DM    2048
#define NH    32
#define NKV   8
#define HD    64

typedef __attribute__((ext_vector_type(16))) __bf16        v16bf;
typedef __attribute__((ext_vector_type(8)))  float         v8f;
typedef __attribute__((ext_vector_type(4)))  unsigned int  v4ui;
typedef __attribute__((ext_vector_type(8)))  int           v8i;
typedef __attribute__((ext_vector_type(4)))  int           v4i;

#ifndef __has_builtin
#define __has_builtin(x) 0
#endif

#if __has_builtin(__builtin_amdgcn_global_load_async_to_lds_b128)
#define HAVE_ASYNC_LDS 1
#else
#define HAVE_ASYNC_LDS 0
#endif

#if __has_builtin(__builtin_amdgcn_tensor_load_to_lds) && \
    __has_builtin(__builtin_amdgcn_s_wait_tensorcnt)
#define HAVE_TDM 1
#else
#define HAVE_TDM 0
#endif

#if __has_builtin(__builtin_amdgcn_s_wait_asynccnt)
#define WAIT_ASYNC() __builtin_amdgcn_s_wait_asynccnt(0)
#else
#define WAIT_ASYNC() asm volatile("s_wait_asynccnt 0x0" ::: "memory")
#endif

// Address-space casts (via-integer pattern, as used by CK's direct-load path).
// The async-to-LDS builtins take (v4i addrspace(1)*, v4i addrspace(3)*, imm, imm).
__device__ inline __attribute__((address_space(1))) v4i* gcast(const void* p) {
  return reinterpret_cast<__attribute__((address_space(1))) v4i*>(
      reinterpret_cast<__SIZE_TYPE__>(p));
}
__device__ inline __attribute__((address_space(3))) v4i* lcast(void* p) {
  return reinterpret_cast<__attribute__((address_space(3))) v4i*>(
      reinterpret_cast<__SIZE_TYPE__>(p));
}

__device__ inline unsigned short f2bf(float f) {      // f32 -> bf16, RNE
  unsigned u = __float_as_uint(f);
  u += 0x7fffu + ((u >> 16) & 1u);
  return (unsigned short)(u >> 16);
}
__device__ inline float bf2f(unsigned short h) {
  return __uint_as_float(((unsigned)h) << 16);
}

// A-matrix fragment (16xK slice, 16-bit elems). Per ISA 7.12.2:
//   lane = half*16 + m ; element e holds k = kbase + 16*(e>>3) + 8*half + (e&7)
// -> two contiguous 16B runs per lane.
__device__ inline v16bf ld_fragA(const unsigned short* lds, int row, int stride,
                                 int kbase, int hl) {
  union { v16bf v; int4 q[2]; } u;
  const unsigned short* p = lds + row * stride + kbase + 8 * hl;
  u.q[0] = *(const int4*)(p);
  u.q[1] = *(const int4*)(p + 16);
  return u.v;
}
// B-matrix fragment (Kx16 slice): lane = half*16 + n ; element e holds
// k = kbase + 16*(e>>3) + 8*half + (e&7)  -> two contiguous 16B runs per lane.
// Source LDS tile must be stored n-major (i.e. B transposed: row n over k).
__device__ inline v16bf ld_fragB(const unsigned short* lds, int row, int stride,
                                 int kbase, int hl) {
  union { v16bf v; int4 q[2]; } u;
  const unsigned short* p = lds + row * stride + kbase + 8 * hl;
  u.q[0] = *(const int4*)(p);
  u.q[1] = *(const int4*)(p + 16);
  return u.v;
}

__device__ inline v8f bf16_mma(v16bf a, v16bf b, v8f c) {
  return __builtin_amdgcn_wmma_f32_16x16x32_bf16(
      /*neg_a=*/false, a, /*neg_b=*/false, b,
      /*c_mod=*/(short)0, c, /*reuse_a=*/false, /*reuse_b=*/false);
}

// ---------------------------------------------------------------------------
// Generic C[M,N] = A[M,K] @ B[K,N].  64x64 block tile, k-step 32, 8 waves,
// each wave owns two 16x16 accumulators sharing one B fragment.
// A: f32 or bf16 (converted to bf16 while staging to LDS). B: f32. C: bf16/f32.
// ---------------------------------------------------------------------------
template <bool A_BF16, bool OUT_F32>
__global__ void __launch_bounds__(256)
gemm_wmma_64x64(const void* __restrict__ Ap, const float* __restrict__ Bp,
                void* __restrict__ Cp, int M, int N, int K) {
  __shared__ unsigned short As[64 * 32];   // m-major [64][32]
  __shared__ unsigned short BsT[64 * 32];  // n-major [64][32] (B transposed)

  const int t = threadIdx.x;
  const int w = t >> 5, lane = t & 31;
  const int hl = lane >> 4, lr = lane & 15;
  const int m0 = blockIdx.y * 64, n0 = blockIdx.x * 64;
  const int mi0 = w >> 2, mi1 = mi0 + 2, ni = w & 3;

  v8f c0 = {0.f, 0.f, 0.f, 0.f, 0.f, 0.f, 0.f, 0.f};
  v8f c1 = {0.f, 0.f, 0.f, 0.f, 0.f, 0.f, 0.f, 0.f};

  const int ea = t * 8;
  const int arow = ea >> 5, acol = ea & 31;  // A tile 64x32
  const int brow = ea >> 6, bcol = ea & 63;  // B tile 32x64

  for (int k0 = 0; k0 < K; k0 += 32) {
    // ---- stage A (convert to bf16 if needed) ----
    if constexpr (A_BF16) {
      const unsigned short* ag =
          (const unsigned short*)Ap + (size_t)(m0 + arow) * K + k0 + acol;
#if HAVE_ASYNC_LDS
      __builtin_amdgcn_global_load_async_to_lds_b128(
          gcast(ag), lcast(&As[arow * 32 + acol]), 0, 0);
#else
      __builtin_prefetch(ag + 32, 0, 1);
      *(int4*)&As[arow * 32 + acol] = *(const int4*)ag;
#endif
    } else {
      const float* ag = (const float*)Ap + (size_t)(m0 + arow) * K + k0 + acol;
      __builtin_prefetch(ag + 32, 0, 1);
      float4 fa = *(const float4*)ag;
      float4 fb = *(const float4*)(ag + 4);
      unsigned p0 = (unsigned)f2bf(fa.x) | ((unsigned)f2bf(fa.y) << 16);
      unsigned p1 = (unsigned)f2bf(fa.z) | ((unsigned)f2bf(fa.w) << 16);
      unsigned p2 = (unsigned)f2bf(fb.x) | ((unsigned)f2bf(fb.y) << 16);
      unsigned p3 = (unsigned)f2bf(fb.z) | ((unsigned)f2bf(fb.w) << 16);
      *(int4*)&As[arow * 32 + acol] = make_int4((int)p0, (int)p1, (int)p2, (int)p3);
    }
    // ---- stage B transposed (n-major) ----
    {
      const float* bg = Bp + (size_t)(k0 + brow) * N + n0 + bcol;
      __builtin_prefetch(bg + (size_t)32 * N, 0, 1);
      float4 f0 = *(const float4*)bg;
      float4 f1 = *(const float4*)(bg + 4);
      float vv[8] = {f0.x, f0.y, f0.z, f0.w, f1.x, f1.y, f1.z, f1.w};
#pragma unroll
      for (int j = 0; j < 8; ++j) BsT[(bcol + j) * 32 + brow] = f2bf(vv[j]);
    }
#if HAVE_ASYNC_LDS
    if constexpr (A_BF16) WAIT_ASYNC();
#endif
    __syncthreads();

    v16bf bf = ld_fragB(BsT, ni * 16 + lr, 32, 0, hl);
    v16bf a0 = ld_fragA(As, mi0 * 16 + lr, 32, 0, hl);
    v16bf a1 = ld_fragA(As, mi1 * 16 + lr, 32, 0, hl);
    c0 = bf16_mma(a0, bf, c0);
    c1 = bf16_mma(a1, bf, c1);
    __syncthreads();
  }

  const int n = n0 + ni * 16 + lr;
#pragma unroll
  for (int r = 0; r < 8; ++r) {
    int ma = m0 + mi0 * 16 + r + 8 * hl;
    int mb = m0 + mi1 * 16 + r + 8 * hl;
    if constexpr (OUT_F32) {
      ((float*)Cp)[(size_t)ma * N + n] = c0[r];
      ((float*)Cp)[(size_t)mb * N + n] = c1[r];
    } else {
      ((unsigned short*)Cp)[(size_t)ma * N + n] = f2bf(c0[r]);
      ((unsigned short*)Cp)[(size_t)mb * N + n] = f2bf(c1[r]);
    }
  }
}

// ---------------------------------------------------------------------------
// RoPE, applied in place to bf16 [B,S,H,64]. One thread per (even,odd) pair.
// Matches reference: emb = concat([freqs,freqs]); out = x*cos(emb)+rot(x)*sin(emb)
// ---------------------------------------------------------------------------
__global__ void rope_kernel(unsigned short* __restrict__ X, int H, int npairs) {
  int idx = blockIdx.x * blockDim.x + threadIdx.x;
  if (idx >= npairs) return;
  int i = idx & 31;           // pair index within head_dim
  int row = idx >> 5;         // (b*S + s)*H + h
  int pos = (row / H) % S_LEN;
  int d0 = 2 * i, d1 = d0 + 1;
  int f0 = (d0 < 32) ? d0 : d0 - 32;
  int f1 = (d1 < 32) ? d1 : d1 - 32;
  const float kLn = 0.28782313662425572f;  // ln(10000)/32
  float a0 = (float)pos * __expf(-(float)f0 * kLn);
  float a1 = (float)pos * __expf(-(float)f1 * kLn);
  size_t base = (size_t)row * 64 + d0;
  float x0 = bf2f(X[base]), x1 = bf2f(X[base + 1]);
  float s0, c0, s1, c1;
  __sincosf(a0, &s0, &c0);
  __sincosf(a1, &s1, &c1);
  X[base]     = f2bf(x0 * c0 - x1 * s0);
  X[base + 1] = f2bf(x1 * c1 + x0 * s1);
}

// ---------------------------------------------------------------------------
// Flash attention: block = (64 query rows, head h, batch b). 8 waves, each
// owns two 16x16 accumulators. QK^T and PV on bf16 WMMA, f32 accumulate,
// online softmax with running max/sum in LDS. Causal mask is analytic.
// K tile via TDM (D# descriptor) when available; Q tile via async-to-LDS.
// ---------------------------------------------------------------------------
__global__ void __launch_bounds__(256)
flash_attn_kernel(const unsigned short* __restrict__ Q,
                  const unsigned short* __restrict__ K,
                  const unsigned short* __restrict__ V,
                  unsigned short* __restrict__ O) {
  __shared__ unsigned short Qs[64 * 64];  // [m][d]
  __shared__ unsigned short Ks[64 * 64];  // [key][d]   (B^T for QK^T)
  __shared__ unsigned short Vt[64 * 64];  // [d][key]   (B^T for PV)
  __shared__ float          Sc[64 * 64];  // f32 score tile
  __shared__ unsigned short Ps[64 * 64];  // bf16 probabilities
  __shared__ float mrun[64], lrun[64], alph[64];

  const int t = threadIdx.x;
  const int w = t >> 5, lane = t & 31;
  const int hl = lane >> 4, lr = lane & 15;
  const int qt = blockIdx.x, h = blockIdx.y, b = blockIdx.z;
  const int kvh = h >> 2;  // NUM_HEADS / NUM_KV_HEADS = 4
  const int q0 = qt * 64;
  const int mi0 = w >> 2, mi1 = mi0 + 2, ni = w & 3;

  if (t < 64) { mrun[t] = -3.0e38f; lrun[t] = 0.f; }

  {  // load Q tile once: rows q0..q0+63 of head h
    const int row = t >> 2, cb = (t & 3) * 16;
    const unsigned short* qg =
        Q + (size_t)(b * S_LEN + q0 + row) * DM + h * HD + cb;
    unsigned short* qdst = &Qs[row * 64 + cb];
#if HAVE_ASYNC_LDS
    __builtin_amdgcn_global_load_async_to_lds_b128(gcast(qg), lcast(qdst), 0, 0);
    __builtin_amdgcn_global_load_async_to_lds_b128(gcast(qg + 8), lcast(qdst + 8), 0, 0);
#else
    *(int4*)qdst       = *(const int4*)qg;
    *(int4*)(qdst + 8) = *(const int4*)(qg + 8);
#endif
  }

  v8f o0 = {0.f, 0.f, 0.f, 0.f, 0.f, 0.f, 0.f, 0.f};
  v8f o1 = {0.f, 0.f, 0.f, 0.f, 0.f, 0.f, 0.f, 0.f};

  for (int jt = 0; jt <= qt; ++jt) {
    const int k0 = jt * 64;
#if HAVE_TDM
    if (w == 0) {  // one wave issues the TDM 2D tile copy: 64 rows x 128B
      const unsigned short* kg =
          K + (size_t)(b * S_LEN + k0) * (NKV * HD) + kvh * HD;
      __SIZE_TYPE__ ga = reinterpret_cast<__SIZE_TYPE__>(kg);
      unsigned lds_a =
          (unsigned)reinterpret_cast<__SIZE_TYPE__>(lcast(&Ks[0]));
      v4ui g0; v8i g1; v4i g2; v4i g3;
      g0[0] = 1u;                                   // count=1 (user D#)
      g0[1] = lds_a;                                // lds_addr
      g0[2] = (unsigned)(ga & 0xffffffffu);         // global_addr[31:0]
      g0[3] = (unsigned)((ga >> 32) & 0x01ffffffu)  // global_addr[56:32]
              | (2u << 30);                         // type=2 ("image")
      g1[0] = (int)(1u << 16);                      // wg_mask=0, data_size=2B
      g1[1] = (int)(((unsigned)(NKV * HD) & 0xffffu) << 16);  // tensor_dim0=512
      g1[2] = (int)(0xffffu << 16);                 // tensor_dim1=0xffff (lo16)
      g1[3] = (int)(64u << 16);                     // tdim1 hi=0 | tile_dim0=64
      g1[4] = 64;                                   // tile_dim1=64, tile_dim2=0
      g1[5] = NKV * HD;                             // tensor_dim0_stride=512
      g1[6] = 0;
      g1[7] = 0;
      g2[0] = 0; g2[1] = 0; g2[2] = 0; g2[3] = 0;   // 2D: groups 2/3 unused
      g3[0] = 0; g3[1] = 0; g3[2] = 0; g3[3] = 0;
#if __clang_major__ >= 23
      __builtin_amdgcn_tensor_load_to_lds(g0, g1, g2, g3, g1, 0);
#else
      __builtin_amdgcn_tensor_load_to_lds(g0, g1, g2, g3, 0);
#endif
      __builtin_amdgcn_s_wait_tensorcnt(0);
    }
#elif HAVE_ASYNC_LDS
    {  // K tile, row-major [key][d] via async copy
      const int row = t >> 2, cb = (t & 3) * 16;
      const unsigned short* kg =
          K + (size_t)(b * S_LEN + k0 + row) * (NKV * HD) + kvh * HD + cb;
      unsigned short* kdst = &Ks[row * 64 + cb];
      __builtin_amdgcn_global_load_async_to_lds_b128(gcast(kg), lcast(kdst), 0, 0);
      __builtin_amdgcn_global_load_async_to_lds_b128(gcast(kg + 8), lcast(kdst + 8), 0, 0);
    }
#else
    {  // K tile, row-major [key][d]
      const int row = t >> 2, cb = (t & 3) * 16;
      const unsigned short* kg =
          K + (size_t)(b * S_LEN + k0 + row) * (NKV * HD) + kvh * HD + cb;
      *(int4*)&Ks[row * 64 + cb]     = *(const int4*)kg;
      *(int4*)&Ks[row * 64 + cb + 8] = *(const int4*)(kg + 8);
    }
#endif
    {  // V tile, stored transposed [d][key] (VALU transpose via LDS stores)
#pragma unroll
      for (int g = 0; g < 2; ++g) {
        int e = t * 16 + g * 8;
        int kr = e >> 6, dc = e & 63;
        const unsigned short* vg =
            V + (size_t)(b * S_LEN + k0 + kr) * (NKV * HD) + kvh * HD + dc;
        union { int4 q; unsigned short u[8]; } tmp;
        tmp.q = *(const int4*)vg;
#pragma unroll
        for (int j = 0; j < 8; ++j) Vt[(dc + j) * 64 + kr] = tmp.u[j];
      }
    }
#if HAVE_ASYNC_LDS
    WAIT_ASYNC();
#endif
    __syncthreads();

    // ---- S = (Q K^T) * scale, causal-masked ----
    v8f s0 = {0.f, 0.f, 0.f, 0.f, 0.f, 0.f, 0.f, 0.f};
    v8f s1 = {0.f, 0.f, 0.f, 0.f, 0.f, 0.f, 0.f, 0.f};
#pragma unroll
    for (int dk = 0; dk < 64; dk += 32) {
      v16bf kb  = ld_fragB(Ks, ni * 16 + lr, 64, dk, hl);
      v16bf qa0 = ld_fragA(Qs, mi0 * 16 + lr, 64, dk, hl);
      v16bf qa1 = ld_fragA(Qs, mi1 * 16 + lr, 64, dk, hl);
      s0 = bf16_mma(qa0, kb, s0);
      s1 = bf16_mma(qa1, kb, s1);
    }
    const int ncol = ni * 16 + lr;
#pragma unroll
    for (int r = 0; r < 8; ++r) {
      int mr  = r + 8 * hl;
      int m0r = mi0 * 16 + mr, m1r = mi1 * 16 + mr;
      Sc[m0r * 64 + ncol] =
          (k0 + ncol <= q0 + m0r) ? s0[r] * 0.125f : -3.0e38f;
      Sc[m1r * 64 + ncol] =
          (k0 + ncol <= q0 + m1r) ? s1[r] * 0.125f : -3.0e38f;
    }
    __syncthreads();

    // ---- online softmax: one thread per query row ----
    if (t < 64) {
      float mt = -3.0e38f;
#pragma unroll 8
      for (int c = 0; c < 64; ++c) mt = fmaxf(mt, Sc[t * 64 + c]);
      float mold = mrun[t];
      float mnew = fmaxf(mold, mt);
      float a = __expf(mold - mnew);
      float sum = 0.f;
#pragma unroll 8
      for (int c = 0; c < 64; ++c) {
        float p = __expf(Sc[t * 64 + c] - mnew);
        sum += p;
        Ps[t * 64 + c] = f2bf(p);
      }
      lrun[t] = lrun[t] * a + sum;
      mrun[t] = mnew;
      alph[t] = a;
    }
    __syncthreads();

    // ---- rescale accumulators, then O += P @ V ----
#pragma unroll
    for (int r = 0; r < 8; ++r) {
      o0[r] *= alph[mi0 * 16 + r + 8 * hl];
      o1[r] *= alph[mi1 * 16 + r + 8 * hl];
    }
#pragma unroll
    for (int kk = 0; kk < 64; kk += 32) {
      v16bf vb = ld_fragB(Vt, ni * 16 + lr, 64, kk, hl);
      v16bf p0 = ld_fragA(Ps, mi0 * 16 + lr, 64, kk, hl);
      v16bf p1 = ld_fragA(Ps, mi1 * 16 + lr, 64, kk, hl);
      o0 = bf16_mma(p0, vb, o0);
      o1 = bf16_mma(p1, vb, o1);
    }
    __syncthreads();  // protect Ks/Vt/Sc/Ps before next key tile
  }

  // ---- epilogue: normalize and store bf16 [B,S,H,64] ----
  const int dcol = ni * 16 + lr;
#pragma unroll
  for (int r = 0; r < 8; ++r) {
    int m0r = mi0 * 16 + r + 8 * hl;
    int m1r = mi1 * 16 + r + 8 * hl;
    float i0 = 1.f / lrun[m0r];
    float i1 = 1.f / lrun[m1r];
    O[(size_t)(b * S_LEN + q0 + m0r) * DM + h * HD + dcol] = f2bf(o0[r] * i0);
    O[(size_t)(b * S_LEN + q0 + m1r) * DM + h * HD + dcol] = f2bf(o1[r] * i1);
  }
}

// ---------------------------------------------------------------------------
extern "C" void kernel_launch(void* const* d_in, const int* in_sizes, int n_in,
                              void* d_out, int out_size, void* d_ws,
                              size_t ws_size, hipStream_t stream) {
  (void)in_sizes; (void)n_in; (void)out_size; (void)ws_size;
  const float* x  = (const float*)d_in[0];
  // d_in[1] is the causal mask; applied analytically in flash_attn_kernel.
  const float* Wq = (const float*)d_in[2];
  const float* Wk = (const float*)d_in[3];
  const float* Wv = (const float*)d_in[4];
  const float* Wo = (const float*)d_in[5];
  float* out = (float*)d_out;

  const int M = 2 * S_LEN;  // B*S = 4096 token rows
  // bf16 workspace: Q(8M) | K(2M) | V(2M) | attn-out(8M) elems = 40 MB
  unsigned short* Qb = (unsigned short*)d_ws;
  unsigned short* Kb = Qb + (size_t)M * DM;
  unsigned short* Vb = Kb + (size_t)M * (NKV * HD);
  unsigned short* Ob = Vb + (size_t)M * (NKV * HD);

  dim3 blk(256);
  // Projections (A=f32 x, C=bf16)
  gemm_wmma_64x64<false, false>
      <<<dim3(DM / 64, M / 64), blk, 0, stream>>>(x, Wq, Qb, M, DM, DM);
  gemm_wmma_64x64<false, false>
      <<<dim3((NKV * HD) / 64, M / 64), blk, 0, stream>>>(x, Wk, Kb, M, NKV * HD, DM);
  gemm_wmma_64x64<false, false>
      <<<dim3((NKV * HD) / 64, M / 64), blk, 0, stream>>>(x, Wv, Vb, M, NKV * HD, DM);
  // RoPE on Q and K (in place)
  int qpairs = M * NH * 32;
  rope_kernel<<<(qpairs + 255) / 256, blk, 0, stream>>>(Qb, NH, qpairs);
  int kpairs = M * NKV * 32;
  rope_kernel<<<(kpairs + 255) / 256, blk, 0, stream>>>(Kb, NKV, kpairs);
  // Flash attention
  flash_attn_kernel<<<dim3(S_LEN / 64, NH, 2), blk, 0, stream>>>(Qb, Kb, Vb, Ob);
  // Output projection (A=bf16 attn-out, C=f32 d_out)
  gemm_wmma_64x64<true, true>
      <<<dim3(DM / 64, M / 64), blk, 0, stream>>>(Ob, Wo, out, M, DM, DM);
}
